// SSWLConv_23184233463959
// MI455X (gfx1250) — matrix operand order repaired and checked by reading, not compiled, not run on hardware
//
#include <hip/hip_runtime.h>

// Problem dims (fixed by the reference)
#define NN   512   // nodes
#define DD   128   // in dim
#define OUTD 128   // out dim
#define K3D  384   // 3*DD

typedef __attribute__((ext_vector_type(16))) __bf16        v16bf;
typedef __attribute__((ext_vector_type(8)))  float         v8f;
typedef __attribute__((ext_vector_type(4)))  unsigned int  u32x4;
typedef __attribute__((ext_vector_type(4)))  float         f32x4;

union Frag16 {
    v16bf  v;
    u32x4  q[2];
    __bf16 h[16];
};

__device__ __forceinline__ v8f wmma_bf16(const Frag16& a, const Frag16& b, v8f c) {
    return __builtin_amdgcn_wmma_f32_16x16x32_bf16(false, a.v, false, b.v,
                                                   (short)0, c, false, false);
}

// A-operand fragment (16x32 tile): u[0..7] = K(kt+half*8+0..7), u[8..15] = K(kt+16+half*8+0..7)
__device__ __forceinline__ void load_afrag(Frag16& f, const __bf16* row, int kt, int half) {
    f.q[0] = *(const u32x4*)(row + kt + half * 8);
    f.q[1] = *(const u32x4*)(row + kt + 16 + half * 8);
}
// B-operand fragment (32x16 tile, column-major source): u[0..7] = K(kt+half*8+0..7), u[8..15] = K(kt+16+half*8+0..7)
__device__ __forceinline__ void load_bfrag(Frag16& f, const __bf16* row, int kt, int half) {
    f.q[0] = *(const u32x4*)(row + kt + half * 8);
    f.q[1] = *(const u32x4*)(row + kt + 16 + half * 8);
}

// ---------------- setup kernels ----------------

__global__ void k_zero(float* A, int n) {
    int t = blockIdx.x * blockDim.x + threadIdx.x;
    if (t < n) A[t] = 0.0f;
}

__global__ void k_scatter(const int* __restrict__ ei, int E, float* __restrict__ A) {
    int t = blockIdx.x * blockDim.x + threadIdx.x;
    if (t < E) {
        int s = ei[t];         // src
        int d = ei[E + t];     // dst
        atomicAdd(A + (size_t)s * NN + d, 1.0f);
    }
}

// ATb[j*NN + k] = bf16(A[k*NN + j])   (transposed adjacency, shared by both aggregations)
__global__ void k_buildAT(const float* __restrict__ A, __bf16* __restrict__ ATb) {
    int t = blockIdx.x * blockDim.x + threadIdx.x;   // t = j*NN + k
    int j = t >> 9;
    int k = t & (NN - 1);
    ATb[t] = (__bf16)A[(size_t)k * NN + j];
}

// WT[o*K3D + c] = bf16(W[c*OUTD + o])  (W transposed so B-fragments are K-contiguous)
__global__ void k_packW(const float* __restrict__ W, __bf16* __restrict__ WT) {
    int t = blockIdx.x * blockDim.x + threadIdx.x;
    if (t < OUTD * K3D) {
        int o = t / K3D;
        int c = t - o * K3D;
        WT[t] = (__bf16)W[(size_t)c * OUTD + o];
    }
}

// ---------------- pass 1: pointwise GEMM ----------------
// rows r = i*NN + j (262144 rows of X, f32 -> bf16 once per block via LDS)
//   out  = X @ W0 + b                              (f32, initializes d_out)
//   Y1T[i][d][k]        = bf16((X @ W1)[i*NN+k, d])   (K-contiguous for pass 2)
//   Y2T[(j*OUTD+d)][k]  = bf16((X @ W2)[k*NN+j, d])   (K-contiguous for pass 3)
__global__ __launch_bounds__(256) void k_pointwise(
    const float* __restrict__ X, const __bf16* __restrict__ WT,
    const float* __restrict__ bias, float* __restrict__ out,
    __bf16* __restrict__ Y1T, __bf16* __restrict__ Y2T)
{
    __shared__ __bf16 xs[16 * DD];                 // 4 KB converted X tile

    const int r0   = blockIdx.x * 16;              // row tile (16 rows, all same i)
    const int t    = threadIdx.x;
    const int wave = t >> 5;                       // 0..7 -> 16-wide output col tile
    const int lane = t & 31;
    const int n    = lane & 15;
    const int half = lane >> 4;
    const int o    = wave * 16 + n;                // output column for this lane

    // cooperative f32 -> bf16 conversion of the 16x128 X tile (8 elements/thread)
    {
        const int row = t >> 4;                    // 0..15
        const int col = (t & 15) * 8;              // 0..120
        const float* xp = X + (size_t)(r0 + row) * DD + col;
        f32x4 f0 = *(const f32x4*)xp;
        f32x4 f1 = *(const f32x4*)(xp + 4);
        union { u32x4 q; __bf16 h[8]; } pk;
#pragma unroll
        for (int e = 0; e < 4; ++e) { pk.h[e] = (__bf16)f0[e]; pk.h[4 + e] = (__bf16)f1[e]; }
        *(u32x4*)(xs + row * DD + col) = pk.q;
    }
    __syncthreads();

    v8f acc0 = {}, acc1 = {}, acc2 = {};
    const __bf16* xrow = xs + n * DD;              // A-frag row = lane%16 (LDS)
    const __bf16* wrow = WT + (size_t)o * K3D;     // B rows = out column o

#pragma unroll
    for (int kt = 0; kt < DD; kt += 32) {
        Frag16 a, b0, b1, b2;
        load_afrag(a, xrow, kt, half);
        load_bfrag(b0, wrow, kt, half);            // W0 block
        load_bfrag(b1, wrow + DD, kt, half);       // W1 block
        load_bfrag(b2, wrow + 2 * DD, kt, half);   // W2 block
        acc0 = wmma_bf16(a, b0, acc0);
        acc1 = wmma_bf16(a, b1, acc1);
        acc2 = wmma_bf16(a, b2, acc2);
    }

    // C/D layout: lane col = n, vgpr v -> row (half*8 + v)
    const int   a0  = r0 >> 9;                     // first axis  (i for Y1, k for Y2)
    const int   b0r = (r0 & (NN - 1)) + half * 8;  // second axis (k for Y1, j for Y2)
    const float bv  = bias[o];

    float* op = out + ((size_t)r0 + half * 8) * OUTD + o;
#pragma unroll
    for (int v = 0; v < 8; ++v) op[(size_t)v * OUTD] = acc0[v] + bv;

    union { u32x4 q; __bf16 h[8]; } y1;
#pragma unroll
    for (int v = 0; v < 8; ++v) y1.h[v] = (__bf16)acc1[v];
    *(u32x4*)(Y1T + ((size_t)a0 * OUTD + o) * NN + b0r) = y1.q;   // 8 contiguous k

#pragma unroll
    for (int v = 0; v < 8; ++v)
        Y2T[((size_t)(b0r + v) * OUTD + o) * NN + a0] = (__bf16)acc2[v];
}

// ---------------- pass 2: within-subgraph aggregation ----------------
// out[i, j-tile, o-tile] += AT[j,:] @ Y1T[i][o][:]   (K = 512)
// Each wave computes a 32x32 output tile (2x2 WMMA tiles, 16 FLOP/B from cache).
__global__ __launch_bounds__(256) void k_aggr_within(
    const __bf16* __restrict__ ATb, const __bf16* __restrict__ Y1T,
    float* __restrict__ out)
{
    const int i     = blockIdx.x >> 3;
    const int j64   = (blockIdx.x & 7) * 64;
    const int wave  = threadIdx.x >> 5;
    const int lane  = threadIdx.x & 31;
    const int n     = lane & 15;
    const int half  = lane >> 4;
    const int jbase = j64 + (wave >> 2) * 32;      // wave's 32 rows (j)
    const int obase = (wave & 3) * 32;             // wave's 32 cols (o)

    const __bf16* arow0 = ATb + (size_t)(jbase + n) * NN;
    const __bf16* arow1 = ATb + (size_t)(jbase + 16 + n) * NN;
    const __bf16* brow0 = Y1T + ((size_t)i * OUTD + obase + n) * NN;
    const __bf16* brow1 = Y1T + ((size_t)i * OUTD + obase + 16 + n) * NN;

    v8f acc00 = {}, acc01 = {}, acc10 = {}, acc11 = {};
    for (int kt = 0; kt < NN; kt += 32) {
        Frag16 a0, a1, b0, b1;
        load_afrag(a0, arow0, kt, half);
        load_afrag(a1, arow1, kt, half);
        load_bfrag(b0, brow0, kt, half);
        load_bfrag(b1, brow1, kt, half);
        acc00 = wmma_bf16(a0, b0, acc00);
        acc01 = wmma_bf16(a0, b1, acc01);
        acc10 = wmma_bf16(a1, b0, acc10);
        acc11 = wmma_bf16(a1, b1, acc11);
    }

    float* base = out + ((size_t)i * NN + jbase + half * 8) * OUTD + obase + n;
#pragma unroll
    for (int v = 0; v < 8; ++v) {
        base[(size_t)v * OUTD]               += acc00[v];
        base[(size_t)v * OUTD + 16]          += acc01[v];
        base[(size_t)(v + 16) * OUTD]        += acc10[v];
        base[(size_t)(v + 16) * OUTD + 16]   += acc11[v];
    }
}

// ---------------- pass 3: cross-subgraph aggregation ----------------
// out[i-tile, j, o-tile] += AT[i,:] @ Y2T[(j,o)][:]   (K = 512), 32x32 per wave
__global__ __launch_bounds__(256) void k_aggr_cross(
    const __bf16* __restrict__ ATb, const __bf16* __restrict__ Y2T,
    float* __restrict__ out)
{
    const int j     = blockIdx.x >> 3;
    const int i64   = (blockIdx.x & 7) * 64;
    const int wave  = threadIdx.x >> 5;
    const int lane  = threadIdx.x & 31;
    const int n     = lane & 15;
    const int half  = lane >> 4;
    const int ibase = i64 + (wave >> 2) * 32;      // wave's 32 rows (i)
    const int obase = (wave & 3) * 32;             // wave's 32 cols (o)

    const __bf16* arow0 = ATb + (size_t)(ibase + n) * NN;
    const __bf16* arow1 = ATb + (size_t)(ibase + 16 + n) * NN;
    const __bf16* brow0 = Y2T + ((size_t)j * OUTD + obase + n) * NN;
    const __bf16* brow1 = Y2T + ((size_t)j * OUTD + obase + 16 + n) * NN;

    v8f acc00 = {}, acc01 = {}, acc10 = {}, acc11 = {};
    for (int kt = 0; kt < NN; kt += 32) {
        Frag16 a0, a1, b0, b1;
        load_afrag(a0, arow0, kt, half);
        load_afrag(a1, arow1, kt, half);
        load_bfrag(b0, brow0, kt, half);
        load_bfrag(b1, brow1, kt, half);
        acc00 = wmma_bf16(a0, b0, acc00);
        acc01 = wmma_bf16(a0, b1, acc01);
        acc10 = wmma_bf16(a1, b0, acc10);
        acc11 = wmma_bf16(a1, b1, acc11);
    }

    // out[((i)*NN + j)*OUTD + o]; i = ibase + {0,16} + half*8 + v
    float* base = out + (((size_t)(ibase + half * 8)) * NN + j) * OUTD + obase + n;
    const size_t si = (size_t)NN * OUTD;           // stride per i
#pragma unroll
    for (int v = 0; v < 8; ++v) {
        base[(size_t)v * si]             += acc00[v];
        base[(size_t)v * si + 16]        += acc01[v];
        base[(size_t)(v + 16) * si]      += acc10[v];
        base[(size_t)(v + 16) * si + 16] += acc11[v];
    }
}

// ---------------- launcher ----------------
extern "C" void kernel_launch(void* const* d_in, const int* in_sizes, int n_in,
                              void* d_out, int out_size, void* d_ws, size_t ws_size,
                              hipStream_t stream)
{
    const float* X    = (const float*)d_in[0];
    const int*   ei   = (const int*)  d_in[1];
    const float* W    = (const float*)d_in[2];
    const float* bias = (const float*)d_in[3];
    float*       out  = (float*)d_out;
    const int    E    = in_sizes[1] / 2;

    // workspace layout (~130 MB total)
    char* ws = (char*)d_ws;
    float*  A   = (float*)ws;                                        // 1 MB
    __bf16* ATb = (__bf16*)(ws + (1u << 20));                        // 0.5 MB
    __bf16* WT  = (__bf16*)(ws + (1u << 20) + (1u << 19));           // 96 KB
    __bf16* Y1T = (__bf16*)(ws + (2u << 20));                        // 64 MB
    __bf16* Y2T = (__bf16*)(ws + (2u << 20) + (64u << 20));          // 64 MB

    k_zero    <<<(NN * NN + 255) / 256, 256, 0, stream>>>(A, NN * NN);
    k_scatter <<<(E + 255) / 256,       256, 0, stream>>>(ei, E, A);
    k_buildAT <<<(NN * NN) / 256,       256, 0, stream>>>(A, ATb);
    k_packW   <<<(OUTD * K3D + 255) / 256, 256, 0, stream>>>(W, WT);

    k_pointwise  <<<(NN * NN) / 16, 256, 0, stream>>>(X, WT, bias, out, Y1T, Y2T);
    k_aggr_within<<<NN * 8,         256, 0, stream>>>(ATb, Y1T, out);
    k_aggr_cross <<<NN * 8,         256, 0, stream>>>(ATb, Y2T, out);
}
